// StandardGPT_1382979469662
// MI455X (gfx1250) — matrix operand order repaired and checked by reading, not compiled, not run on hardware
//
#include <hip/hip_runtime.h>
#include <hip/hip_bf16.h>

typedef __bf16 bf16;
typedef __attribute__((ext_vector_type(16))) __bf16 v16bf;
typedef __attribute__((ext_vector_type(8)))  float  v8f;

namespace {
constexpr int Bb = 4, Tt = 1024, Cc = 768, NHh = 12, HSz = 64, NLl = 12, Vv = 32000;
constexpr int Mm = Bb * Tt;     // 4096 rows
constexpr int C3 = 3 * Cc;      // 2304
constexpr int C4 = 4 * Cc;      // 3072
constexpr int LSTR = 40;        // padded LDS row stride (elems): 80B = 20 banks, conflict-free
}

// ---------------- WMMA fragment helpers (CDNA5 16x16x32 bf16) ----------------
// A 16x32 (16-bit): lanes 0-15 row M=lane, VGPR0-3 = K 0..7, VGPR4-7 = K 16..23
//                   lanes 16-31 row M=lane-16, VGPR0-3 = K 8..15, VGPR4-7 = K 24..31
static __device__ __forceinline__ v16bf load_frag_a(const bf16* base, int stride, int lane) {
    int row = lane & 15;
    int ka  = (lane >> 4) << 3;               // 0 or 8
    const bf16* p = base + (long)row * stride + ka;
    v16bf a;
    reinterpret_cast<float4*>(&a)[0] = *reinterpret_cast<const float4*>(p);
    reinterpret_cast<float4*>(&a)[1] = *reinterpret_cast<const float4*>(p + 16);
    return a;
}
// B 32x16 (16-bit), from Bt stored as [N][K] row-major: lane n holds column n.
// lanes 0-15: K 0..7, 16..23; lanes 16-31: K 8..15, 24..31 (same as A).
static __device__ __forceinline__ v16bf load_frag_b(const bf16* base, int stride, int lane) {
    int row = lane & 15;
    int kb  = (lane >> 4) << 3;               // 0 or 8
    const bf16* p = base + (long)row * stride + kb;
    v16bf b;
    reinterpret_cast<float4*>(&b)[0] = *reinterpret_cast<const float4*>(p);
    reinterpret_cast<float4*>(&b)[1] = *reinterpret_cast<const float4*>(p + 16);
    return b;
}
static __device__ __forceinline__ v8f wmma_bf16(v16bf a, v16bf b, v8f c) {
    return __builtin_amdgcn_wmma_f32_16x16x32_bf16(false, a, false, b, (short)0, c, false, false);
}

// CDNA5 async DMA: 16B global -> LDS, tracked by ASYNCcnt (no VGPR round trip).
// LDS dest VGPR = low 32 bits of flat pointer (= LDS byte offset per aperture rules).
static __device__ __forceinline__ void async_cp16(bf16* ldsDst, const bf16* gSrc) {
    unsigned l = (unsigned)(size_t)ldsDst;
    asm volatile("global_load_async_to_lds_b128 %0, %1, off"
                 :: "v"(l), "v"(gSrc) : "memory");
}

// ---------------- fp32 -> bf16 transposed weight conversion ----------------
// src: [K][N] fp32 (+ z*srcZ), dst: [N][K] bf16 (+ z*dstZ)
__global__ __launch_bounds__(256) void transpose_conv_kernel(
    const float* __restrict__ src, bf16* __restrict__ dst,
    int K, int N, long srcZ, long dstZ) {
    __shared__ float tile[32][33];
    src += (long)blockIdx.z * srcZ;
    dst += (long)blockIdx.z * dstZ;
    int n0 = blockIdx.x * 32, k0 = blockIdx.y * 32;
    int tx = threadIdx.x & 31, ty = threadIdx.x >> 5;
#pragma unroll
    for (int j = 0; j < 4; ++j)
        tile[ty + 8 * j][tx] = src[(long)(k0 + ty + 8 * j) * N + n0 + tx];
    __syncthreads();
#pragma unroll
    for (int j = 0; j < 4; ++j)
        dst[(long)(n0 + ty + 8 * j) * K + k0 + tx] = (bf16)tile[tx][ty + 8 * j];
}

// ---------------- V transpose: qkv[t][2C + h*64 + d] -> Vt[(b,h)][d][t] ----------------
__global__ __launch_bounds__(256) void transpose_v_kernel(
    const bf16* __restrict__ qkv, bf16* __restrict__ Vt) {
    __shared__ bf16 tile[32][34];
    int z = blockIdx.z;                        // b*12 + h
    int b = z / NHh, h = z % NHh;
    const bf16* src = qkv + (long)(b * Tt) * C3 + 2 * Cc + h * HSz;   // [1024][64], ld=C3
    bf16* dst = Vt + (long)z * HSz * Tt;                              // [64][1024]
    int t0 = blockIdx.y * 32, d0 = blockIdx.x * 32;
    int tx = threadIdx.x & 31, ty = threadIdx.x >> 5;
#pragma unroll
    for (int j = 0; j < 4; ++j)
        tile[ty + 8 * j][tx] = src[(long)(t0 + ty + 8 * j) * C3 + d0 + tx];
    __syncthreads();
#pragma unroll
    for (int j = 0; j < 4; ++j)
        dst[(long)(d0 + ty + 8 * j) * Tt + t0 + tx] = tile[tx][ty + 8 * j];
}

// ---------------- embedding ----------------
__global__ __launch_bounds__(256) void embed_kernel(
    const int* __restrict__ idx, const float* __restrict__ tok,
    const float* __restrict__ pos, float* __restrict__ x) {
    long i = (long)blockIdx.x * 256 + threadIdx.x;   // Mm*Cc total
    int row = (int)(i / Cc), c = (int)(i % Cc);
    int t = row % Tt;
    x[i] = tok[(long)idx[row] * Cc + c] + pos[(long)t * Cc + c];
}

// ---------------- LayerNorm (biased var, eps 1e-5), fp32 -> bf16 ----------------
__global__ __launch_bounds__(256) void layernorm_kernel(
    const float* __restrict__ x, const float* __restrict__ g,
    const float* __restrict__ be, bf16* __restrict__ out) {
    int row = blockIdx.x;
    const float* xr = x + (long)row * Cc;
    int tid = threadIdx.x, lane = tid & 31, wid = tid >> 5;
    float v[3], s = 0.f, ss = 0.f;
#pragma unroll
    for (int i = 0; i < 3; ++i) { v[i] = xr[tid + 256 * i]; s += v[i]; ss += v[i] * v[i]; }
#pragma unroll
    for (int off = 16; off >= 1; off >>= 1) {
        s  += __shfl_xor(s, off, 32);
        ss += __shfl_xor(ss, off, 32);
    }
    __shared__ float red[16];
    if (lane == 0) { red[wid] = s; red[8 + wid] = ss; }
    __syncthreads();
    float ts = 0.f, tss = 0.f;
#pragma unroll
    for (int w = 0; w < 8; ++w) { ts += red[w]; tss += red[8 + w]; }
    float mean = ts * (1.0f / Cc);
    float var = tss * (1.0f / Cc) - mean * mean;
    float rstd = rsqrtf(var + 1e-5f);
#pragma unroll
    for (int i = 0; i < 3; ++i) {
        int c = tid + 256 * i;
        out[(long)row * Cc + c] = (bf16)((v[i] - mean) * rstd * g[c] + be[c]);
    }
}

// ---------------- WMMA GEMM: out[M][N] = A[M][K](bf16) * Bt[N][K]^T ----------------
// Block: 128(M) x 256(N), BK=32, 8 waves in 2x4 grid, wave tile 64x64 (16 WMMA/k-step).
// Async double-buffered LDS staging (GLOBAL_LOAD_ASYNC_TO_LDS_B128 + s_wait_asynccnt).
enum { MODE_BF16 = 0, MODE_GELU = 1, MODE_F32_RES = 2, MODE_F32 = 3 };

static __device__ __forceinline__ void stage_tile_async(
    const bf16* __restrict__ A, int lda, const bf16* __restrict__ Bt, int ldb,
    bf16* As, bf16* Bs, int m0, int n0, int kt, int tid) {
#pragma unroll
    for (int j = 0; j < 2; ++j) {               // A tile: 128 x 32
        int idx = tid + 256 * j;
        int row = idx >> 2, kc = (idx & 3) << 3;
        async_cp16(As + row * LSTR + kc, A + (long)(m0 + row) * lda + kt + kc);
    }
#pragma unroll
    for (int j = 0; j < 4; ++j) {               // B tile: 256 x 32
        int idx = tid + 256 * j;
        int row = idx >> 2, kc = (idx & 3) << 3;
        async_cp16(Bs + row * LSTR + kc, Bt + (long)(n0 + row) * ldb + kt + kc);
    }
}

template <int MODE>
__global__ __launch_bounds__(256) void gemm_nt_kernel(
    const bf16* __restrict__ A, int lda, const bf16* __restrict__ Bt, int ldb,
    const float* __restrict__ bias, float* __restrict__ outf,
    bf16* __restrict__ outh, int ldo, int K) {
    __shared__ __align__(16) bf16 As[2][128 * LSTR];
    __shared__ __align__(16) bf16 Bs[2][256 * LSTR];
    const int tid = threadIdx.x, lane = tid & 31, wid = tid >> 5;
    const int wm = wid & 1, wn = wid >> 1;          // 2 x 4 wave grid
    const int m0 = blockIdx.y * 128, n0 = blockIdx.x * 256;

    v8f acc[4][4];
#pragma unroll
    for (int mi = 0; mi < 4; ++mi)
#pragma unroll
        for (int ni = 0; ni < 4; ++ni) acc[mi][ni] = 0.f;

    const int nk = K >> 5;
    stage_tile_async(A, lda, Bt, ldb, As[0], Bs[0], m0, n0, 0, tid);

    for (int t = 0; t < nk; ++t) {
        const int buf = t & 1;
        if (t + 1 < nk) {
            stage_tile_async(A, lda, Bt, ldb, As[buf ^ 1], Bs[buf ^ 1],
                             m0, n0, (t + 1) << 5, tid);
            asm volatile("s_wait_asynccnt 6" ::: "memory");   // tile t landed
        } else {
            asm volatile("s_wait_asynccnt 0" ::: "memory");
        }
        __syncthreads();
        v16bf bfr[4];
#pragma unroll
        for (int ni = 0; ni < 4; ++ni)
            bfr[ni] = load_frag_b(&Bs[buf][(wn * 64 + ni * 16) * LSTR], LSTR, lane);
#pragma unroll
        for (int mi = 0; mi < 4; ++mi) {
            v16bf afr = load_frag_a(&As[buf][(wm * 64 + mi * 16) * LSTR], LSTR, lane);
#pragma unroll
            for (int ni = 0; ni < 4; ++ni)
                acc[mi][ni] = wmma_bf16(afr, bfr[ni], acc[mi][ni]);
        }
        __syncthreads();
    }

    const int cn = lane & 15, hi = lane >> 4;
#pragma unroll
    for (int mi = 0; mi < 4; ++mi)
#pragma unroll
        for (int ni = 0; ni < 4; ++ni) {
            int gm = m0 + wm * 64 + mi * 16 + hi * 8;
            int gn = n0 + wn * 64 + ni * 16 + cn;
            float bv = (MODE == MODE_BF16) ? 0.f : bias[gn];
#pragma unroll
            for (int r = 0; r < 8; ++r) {
                float val = acc[mi][ni][r];
                long o = (long)(gm + r) * ldo + gn;
                if (MODE == MODE_BF16) {
                    outh[o] = (bf16)val;
                } else if (MODE == MODE_GELU) {
                    val += bv;
                    val = 0.5f * val * (1.0f + erff(val * 0.70710678118654752f));
                    outh[o] = (bf16)val;
                } else if (MODE == MODE_F32_RES) {
                    outf[o] += val + bv;          // residual already in outf
                } else {
                    outf[o] = val + bv;
                }
            }
        }
}

// ---------------- Flash attention: one wave per (b, h, 16 q-rows) ----------------
__global__ __launch_bounds__(256) void attention_kernel(
    const bf16* __restrict__ qkv, const bf16* __restrict__ Vt,
    bf16* __restrict__ out) {
    __shared__ __align__(16) bf16 Pbuf[8][16 * 32];
    const int tid = threadIdx.x, lane = tid & 31, wid = tid >> 5;
    const int u = blockIdx.x * 8 + wid;            // 0..3071
    const int qb = u & 63, h = (u >> 6) % NHh, b = u / (64 * NHh);
    const int q0 = qb * 16;
    const float scale = 0.036084391824351614f;     // 768^-0.5 (full C, per reference)
    const int ldq = C3;
    const bf16* qbase = qkv + (long)(b * Tt) * ldq + h * HSz;
    const bf16* kbase = qkv + (long)(b * Tt) * ldq + Cc + h * HSz;
    const bf16* vbase = Vt + (long)((b * NHh + h) * HSz) * Tt;
    const int row16 = lane & 15, hi = lane >> 4;

    v16bf qa0 = load_frag_a(qbase + (long)q0 * ldq, ldq, lane);
    v16bf qa1 = load_frag_a(qbase + (long)q0 * ldq + 32, ldq, lane);

    v8f o0 = 0.f, o1 = 0.f, o2 = 0.f, o3 = 0.f;
    float m[8], l[8];
#pragma unroll
    for (int r = 0; r < 8; ++r) { m[r] = -1e30f; l[r] = 0.f; }
    bf16* Pw = Pbuf[wid];

    for (int kv0 = 0; kv0 <= q0 + 15; kv0 += 32) {
        v8f s[2];
#pragma unroll
        for (int st = 0; st < 2; ++st) {
            const bf16* kb = kbase + (long)(kv0 + 16 * st) * ldq;
            v16bf b0 = load_frag_b(kb, ldq, lane);       // d 0..31
            v16bf b1 = load_frag_b(kb + 32, ldq, lane);  // d 32..63
            v8f t = 0.f;
            t = wmma_bf16(qa0, b0, t);
            t = wmma_bf16(qa1, b1, t);
            int kj = kv0 + 16 * st + row16;
#pragma unroll
            for (int r = 0; r < 8; ++r) {
                float v = t[r] * scale;
                int qi = q0 + hi * 8 + r;
                if (kj > qi) v = -1e30f;                  // causal mask
                t[r] = v;
            }
            s[st] = t;
        }
        float alpha[8];
#pragma unroll
        for (int r = 0; r < 8; ++r) {
            float v = fmaxf(s[0][r], s[1][r]);
#pragma unroll
            for (int off = 8; off >= 1; off >>= 1) v = fmaxf(v, __shfl_xor(v, off, 16));
            float mn = fmaxf(m[r], v);
            alpha[r] = __expf(m[r] - mn);
            m[r] = mn;
        }
#pragma unroll
        for (int st = 0; st < 2; ++st)
#pragma unroll
            for (int r = 0; r < 8; ++r) {
                float p = __expf(s[st][r] - m[r]);
                s[st][r] = p;
                Pw[(hi * 8 + r) * 32 + 16 * st + row16] = (bf16)p;
            }
#pragma unroll
        for (int r = 0; r < 8; ++r) {
            float v = s[0][r] + s[1][r];
#pragma unroll
            for (int off = 8; off >= 1; off >>= 1) v += __shfl_xor(v, off, 16);
            l[r] = l[r] * alpha[r] + v;
            o0[r] *= alpha[r]; o1[r] *= alpha[r]; o2[r] *= alpha[r]; o3[r] *= alpha[r];
        }
        asm volatile("s_wait_dscnt 0" ::: "memory");      // in-wave LDS W->R ordering
        v16bf pa = load_frag_a(Pw, 32, lane);             // P 16x32 as A-fragment
        v16bf v0 = load_frag_b(vbase + kv0, Tt, lane);
        v16bf v1 = load_frag_b(vbase + 16 * Tt + kv0, Tt, lane);
        v16bf v2 = load_frag_b(vbase + 32 * Tt + kv0, Tt, lane);
        v16bf v3 = load_frag_b(vbase + 48 * Tt + kv0, Tt, lane);
        o0 = wmma_bf16(pa, v0, o0);
        o1 = wmma_bf16(pa, v1, o1);
        o2 = wmma_bf16(pa, v2, o2);
        o3 = wmma_bf16(pa, v3, o3);
    }

    bf16* ob = out + (long)(b * Tt + q0) * Cc + h * HSz;
#pragma unroll
    for (int r = 0; r < 8; ++r) {
        float inv = 1.0f / l[r];
        long rr = (long)(hi * 8 + r) * Cc;
        ob[rr + 0  + row16] = (bf16)(o0[r] * inv);
        ob[rr + 16 + row16] = (bf16)(o1[r] * inv);
        ob[rr + 32 + row16] = (bf16)(o2[r] * inv);
        ob[rr + 48 + row16] = (bf16)(o3[r] * inv);
    }
}

// ---------------- host orchestration ----------------
extern "C" void kernel_launch(void* const* d_in, const int* in_sizes, int n_in,
                              void* d_out, int out_size, void* d_ws, size_t ws_size,
                              hipStream_t stream) {
    const int*   idx   = (const int*)d_in[0];
    const float* tok   = (const float*)d_in[1];
    const float* pos   = (const float*)d_in[2];
    const float* Wq    = (const float*)d_in[3];
    const float* Wk    = (const float*)d_in[4];
    const float* Wv    = (const float*)d_in[5];
    const float* Wproj = (const float*)d_in[6];
    const float* bproj = (const float*)d_in[7];
    const float* ln1g  = (const float*)d_in[8];
    const float* ln1b  = (const float*)d_in[9];
    const float* ln2g  = (const float*)d_in[10];
    const float* ln2b  = (const float*)d_in[11];
    const float* W1    = (const float*)d_in[12];
    const float* b1    = (const float*)d_in[13];
    const float* W2    = (const float*)d_in[14];
    const float* b2    = (const float*)d_in[15];
    const float* lnfg  = (const float*)d_in[16];
    const float* lnfb  = (const float*)d_in[17];
    const float* Whead = (const float*)d_in[18];
    const float* bhead = (const float*)d_in[19];

    char* wsp = (char*)d_ws;
    auto alloc = [&](size_t bytes) -> char* {
        char* p = wsp; wsp += (bytes + 255) & ~(size_t)255; return p;
    };
    bf16* qkvT  = (bf16*)alloc((size_t)NLl * C3 * Cc * 2);   // [L][3C][C]
    bf16* projT = (bf16*)alloc((size_t)NLl * Cc * Cc * 2);   // [L][C][C]
    bf16* w1T   = (bf16*)alloc((size_t)NLl * C4 * Cc * 2);   // [L][4C][C]
    bf16* w2T   = (bf16*)alloc((size_t)NLl * Cc * C4 * 2);   // [L][C][4C]
    bf16* headT = (bf16*)alloc((size_t)Vv * Cc * 2);         // [V][C]
    float* x    = (float*)alloc((size_t)Mm * Cc * 4);
    bf16* hbuf  = (bf16*)alloc((size_t)Mm * Cc * 2);
    bf16* qkv   = (bf16*)alloc((size_t)Mm * C3 * 2);
    bf16* VtB   = (bf16*)alloc((size_t)Mm * Cc * 2);
    bf16* att   = (bf16*)alloc((size_t)Mm * Cc * 2);
    bf16* ubuf  = (bf16*)alloc((size_t)Mm * C4 * 2);
    (void)in_sizes; (void)n_in; (void)out_size; (void)ws_size;

    dim3 blk(256);
    long CC = (long)Cc * Cc, C4C = (long)Cc * C4, qz = (long)C3 * Cc;
    // weight conversion + transpose (every call; deterministic)
    transpose_conv_kernel<<<dim3(Cc/32, Cc/32, NLl), blk, 0, stream>>>(Wq,    qkvT,        Cc, Cc, CC,  qz);
    transpose_conv_kernel<<<dim3(Cc/32, Cc/32, NLl), blk, 0, stream>>>(Wk,    qkvT + CC,   Cc, Cc, CC,  qz);
    transpose_conv_kernel<<<dim3(Cc/32, Cc/32, NLl), blk, 0, stream>>>(Wv,    qkvT + 2*CC, Cc, Cc, CC,  qz);
    transpose_conv_kernel<<<dim3(Cc/32, Cc/32, NLl), blk, 0, stream>>>(Wproj, projT,       Cc, Cc, CC,  CC);
    transpose_conv_kernel<<<dim3(C4/32, Cc/32, NLl), blk, 0, stream>>>(W1,    w1T,         Cc, C4, C4C, C4C);
    transpose_conv_kernel<<<dim3(Cc/32, C4/32, NLl), blk, 0, stream>>>(W2,    w2T,         C4, Cc, C4C, C4C);
    transpose_conv_kernel<<<dim3(Vv/32, Cc/32, 1),   blk, 0, stream>>>(Whead, headT,       Cc, Vv, 0,   0);

    embed_kernel<<<Mm * Cc / 256, blk, 0, stream>>>(idx, tok, pos, x);

    for (int layer = 0; layer < NLl; ++layer) {
        layernorm_kernel<<<Mm, blk, 0, stream>>>(x, ln1g + layer*Cc, ln1b + layer*Cc, hbuf);
        gemm_nt_kernel<MODE_BF16><<<dim3(C3/256, Mm/128), blk, 0, stream>>>(
            hbuf, Cc, qkvT + (long)layer*qz, Cc, nullptr, nullptr, qkv, C3, Cc);
        transpose_v_kernel<<<dim3(HSz/32, Tt/32, Bb*NHh), blk, 0, stream>>>(qkv, VtB);
        attention_kernel<<<Bb*NHh*(Tt/16)/8, blk, 0, stream>>>(qkv, VtB, att);
        gemm_nt_kernel<MODE_F32_RES><<<dim3(Cc/256, Mm/128), blk, 0, stream>>>(
            att, Cc, projT + (long)layer*CC, Cc, bproj + layer*Cc, x, nullptr, Cc, Cc);
        layernorm_kernel<<<Mm, blk, 0, stream>>>(x, ln2g + layer*Cc, ln2b + layer*Cc, hbuf);
        gemm_nt_kernel<MODE_GELU><<<dim3(C4/256, Mm/128), blk, 0, stream>>>(
            hbuf, Cc, w1T + (long)layer*C4C, Cc, b1 + layer*C4, nullptr, ubuf, C4, Cc);
        gemm_nt_kernel<MODE_F32_RES><<<dim3(Cc/256, Mm/128), blk, 0, stream>>>(
            ubuf, C4, w2T + (long)layer*C4C, C4, b2 + layer*Cc, x, nullptr, Cc, C4);
    }

    layernorm_kernel<<<Mm, blk, 0, stream>>>(x, lnfg, lnfb, hbuf);
    gemm_nt_kernel<MODE_F32><<<dim3(Vv/256, Mm/128), blk, 0, stream>>>(
        hbuf, Cc, headT, Cc, bhead, (float*)d_out, nullptr, Vv, Cc);
}
